// ResBlock_16913581212018
// MI455X (gfx1250) — compile-verified, matrix operand order checked by reading, not run on hardware
//
#include <hip/hip_runtime.h>
#include <hip/hip_bf16.h>

typedef __attribute__((ext_vector_type(16))) __bf16 v16bf;
typedef __attribute__((ext_vector_type(8)))  __bf16 v8bf;
typedef __attribute__((ext_vector_type(8)))  float  v8f;

// ---------- helpers ----------

static __device__ __forceinline__ __bf16 f2b(float f) {
    union { float f; unsigned u; } in; in.f = f;
    unsigned u = in.u + 0x7FFFu + ((in.u >> 16) & 1u);   // round-to-nearest-even
    union { unsigned short s; __bf16 b; } out; out.s = (unsigned short)(u >> 16);
    return out.b;
}

// Load a 16x32 bf16 WMMA A/B fragment from a row-major matrix.
// Per ISA layout: lane l (l<16) row=l holds K = {k0..k0+7, k0+16..k0+23};
// lanes 16..31 hold the same rows with K shifted by +8.  Both K runs are
// contiguous in memory for row-major data -> two 16-byte loads per lane.
static __device__ __forceinline__ v16bf ld_frag(const __bf16* __restrict__ base,
                                                size_t row0, int k0, int ld) {
    const int lane = threadIdx.x & 31;
    const __bf16* p = base + (row0 + (size_t)(lane & 15)) * (size_t)ld
                           + k0 + ((lane >> 4) << 3);
    v8bf lo = *(const v8bf*)p;
    v8bf hi = *(const v8bf*)(p + 16);
    return __builtin_shufflevector(lo, hi, 0,1,2,3,4,5,6,7,8,9,10,11,12,13,14,15);
}

static __device__ __forceinline__ v8f wmma_bf16(v16bf a, v16bf b, v8f c) {
    return __builtin_amdgcn_wmma_f32_16x16x32_bf16(false, a, false, b,
                                                   (short)0, c, false, false);
}

// ---------- elementwise / setup kernels ----------

__global__ void zero_kernel(float* __restrict__ p, int n) {
    int i = blockIdx.x * blockDim.x + threadIdx.x;
    if (i < n) p[i] = 0.f;
}

__global__ void f2b_kernel(const float* __restrict__ in, __bf16* __restrict__ out, int n) {
    int i = blockIdx.x * blockDim.x + threadIdx.x;
    if (i < n) out[i] = f2b(in[i]);
}

// Expand C[8,32,32] into a 256x256 block-diagonal bf16 matrix (row=h*32+j, col=h*32+k)
__global__ void cblk_kernel(const float* __restrict__ C, __bf16* __restrict__ out) {
    int i = blockIdx.x * 256 + threadIdx.x;          // 65536 elements
    int row = i >> 8, col = i & 255;
    int h = row >> 5;
    float v = ((col >> 5) == h) ? C[h * 1024 + (row & 31) * 32 + (col & 31)] : 0.f;
    out[i] = f2b(v);
}

// XpT[(b*8+h)*32 + k][n] = Xp[b*1024+n][h*32+k]   (bf16, out-index centric)
__global__ void transpose_kernel(const __bf16* __restrict__ Xp, __bf16* __restrict__ XpT) {
    size_t idx = (size_t)blockIdx.x * blockDim.x + threadIdx.x;
    if (idx >= 4194304ull) return;
    int nn = (int)(idx & 1023);
    int k  = (int)((idx >> 10) & 31);
    int bh = (int)(idx >> 15);
    int b = bh >> 3, h = bh & 7;
    XpT[idx] = Xp[(((size_t)b << 10) + nn) * 256 + h * 32 + k];
}

// ---------- generic 16384x256x256 GEMM:  Out = act(In @ W^T + bias) ----------
// mode 0: store f32      mode 1: store bf16
// mode 2: x2 = acc+bias; g = sigmoid(x1+x2); store x1*g + x2*(1-g)  (f32)
__global__ void __launch_bounds__(128) gemm256_kernel(
        const __bf16* __restrict__ In, const __bf16* __restrict__ W,
        const float* __restrict__ bias, const float* __restrict__ x1buf,
        float* __restrict__ outF, __bf16* __restrict__ outB, int mode)
{
    const int m0   = blockIdx.x * 16;
    const int wave = threadIdx.x >> 5;
    const int lane = threadIdx.x & 31;
    const int n0   = wave * 64;
    const v8f vz = {0.f,0.f,0.f,0.f,0.f,0.f,0.f,0.f};
    v8f acc[4] = {vz, vz, vz, vz};
    for (int kc = 0; kc < 256; kc += 32) {
        v16bf a = ld_frag(In, (size_t)m0, kc, 256);
        #pragma unroll
        for (int t = 0; t < 4; ++t) {
            v16bf b = ld_frag(W, (size_t)(n0 + t * 16), kc, 256);
            acc[t] = wmma_bf16(a, b, acc[t]);
        }
    }
    const int n  = lane & 15;
    const int mo = (lane >> 4) << 3;
    #pragma unroll
    for (int t = 0; t < 4; ++t) {
        int col = n0 + t * 16 + n;
        float bv = bias ? bias[col] : 0.f;
        #pragma unroll
        for (int r = 0; r < 8; ++r) {
            size_t idx = (size_t)(m0 + mo + r) * 256 + col;
            float v = acc[t][r] + bv;
            if (mode == 0) {
                outF[idx] = v;
            } else if (mode == 1) {
                outB[idx] = f2b(v);
            } else {
                float a1 = x1buf[idx];
                float g  = 1.f / (1.f + __expf(-(a1 + v)));
                outF[idx] = a1 * g + v * (1.f - g);
            }
        }
    }
}

// ---------- fused masked attention ----------
// grid (N/16, B), block 256 (8 waves, one head each).  Mask tile shared in LDS.
__global__ void __launch_bounds__(256) attn_kernel(
        const __bf16* __restrict__ Xp, const __bf16* __restrict__ Xc,
        const __bf16* __restrict__ XpT, const float* __restrict__ Amask,
        float* __restrict__ xattn)
{
    __shared__ __align__(16) float  maskT[16 * 32];       // 2 KB, shared by all heads
    __shared__ __align__(16) __bf16 sstage[8][16 * 32];   // per-wave S staging
    const int b    = blockIdx.y;
    const int r0   = blockIdx.x * 16;
    const int wave = threadIdx.x >> 5;                    // head index
    const int lane = threadIdx.x & 31;
    const size_t rowbase = (size_t)b << 10;               // b*N
    const __bf16* Xph  = Xp + wave * 32;
    const __bf16* Xch  = Xc + wave * 32;
    const __bf16* XpTh = XpT + ((size_t)(b * 8 + wave) * 32) * 1024;

    const v8f vz = {0.f,0.f,0.f,0.f,0.f,0.f,0.f,0.f};
    v16bf a_xc = ld_frag(Xch, rowbase + r0, 0, 256);      // query rows, K=AD=32
    v8f acc0 = vz, acc1 = vz;
    const int n  = lane & 15;
    const int mo = (lane >> 4) << 3;

    for (int m0 = 0; m0 < 1024; m0 += 32) {
        __syncthreads();                                   // protect maskT readers
        for (int i = threadIdx.x; i < 512; i += 256) {
            int rr = i >> 5, cc = i & 31;
            maskT[i] = Amask[((rowbase + r0 + rr) << 10) + m0 + cc];
        }
        __syncthreads();
        // S = Xc_tile @ Xp_block^T   (two 16x16 tiles, K=32)
        v16bf bk0 = ld_frag(Xph, rowbase + m0,      0, 256);
        v16bf bk1 = ld_frag(Xph, rowbase + m0 + 16, 0, 256);
        v8f s0 = wmma_bf16(a_xc, bk0, vz);
        v8f s1 = wmma_bf16(a_xc, bk1, vz);
        // mask * tanh, stage as bf16 [16 q][32 m] for the second matmul
        #pragma unroll
        for (int r = 0; r < 8; ++r) {
            int qr = mo + r;
            sstage[wave][qr * 32 + n]      = f2b(tanhf(maskT[qr * 32 + n]      * s0[r]));
            sstage[wave][qr * 32 + 16 + n] = f2b(tanhf(maskT[qr * 32 + 16 + n] * s1[r]));
        }
        // Xh += S @ Xp_block   (K = 32 keys; B fragment from pre-transposed XpT)
        v16bf a_s = ld_frag(&sstage[wave][0], 0, 0, 32);   // same-wave LDS RAW: in-order
        v16bf bv0 = ld_frag(XpTh,                     0, m0, 1024);
        v16bf bv1 = ld_frag(XpTh + (size_t)16 * 1024, 0, m0, 1024);
        acc0 = wmma_bf16(a_s, bv0, acc0);
        acc1 = wmma_bf16(a_s, bv1, acc1);
    }
    #pragma unroll
    for (int r = 0; r < 8; ++r) {
        size_t row = rowbase + r0 + mo + r;
        xattn[row * 256 + wave * 32 + n]      = fmaxf(acc0[r], 0.f);
        xattn[row * 256 + wave * 32 + 16 + n] = fmaxf(acc1[r], 0.f);
    }
}

// ---------- BatchNorm (training-mode, biased variance) ----------

__global__ void __launch_bounds__(256) bnstats_kernel(const float* __restrict__ x,
                                                      float* __restrict__ sum,
                                                      float* __restrict__ sumsq) {
    int ch = threadIdx.x;
    size_t row0 = (size_t)blockIdx.x * 256;
    float s = 0.f, ss = 0.f;
    for (int r = 0; r < 256; ++r) {
        float v = x[(row0 + r) * 256 + ch];
        s += v; ss += v * v;
    }
    atomicAdd(&sum[ch], s);
    atomicAdd(&sumsq[ch], ss);
}

__global__ void bnapply_kernel(const float* __restrict__ x,
                               const float* __restrict__ sum, const float* __restrict__ sumsq,
                               const float* __restrict__ gamma, const float* __restrict__ beta,
                               float* __restrict__ outF, __bf16* __restrict__ outB) {
    size_t i = (size_t)blockIdx.x * blockDim.x + threadIdx.x;
    if (i >= 4194304ull) return;
    int ch = (int)(i & 255);
    float m   = sum[ch]   * (1.f / 16384.f);
    float var = sumsq[ch] * (1.f / 16384.f) - m * m;
    float v = (x[i] - m) * rsqrtf(var + 1e-5f) * gamma[ch] + beta[ch];
    v = fmaxf(v, 0.f);
    if (outB) outB[i] = f2b(v);
    else      outF[i] = v;
}

// ---------- launcher ----------

extern "C" void kernel_launch(void* const* d_in, const int* in_sizes, int n_in,
                              void* d_out, int out_size, void* d_ws, size_t ws_size,
                              hipStream_t stream) {
    (void)in_sizes; (void)n_in; (void)out_size; (void)ws_size;
    const float* X   = (const float*)d_in[0];
    const float* A   = (const float*)d_in[1];
    const float* Wp  = (const float*)d_in[2];
    const float* bp  = (const float*)d_in[3];
    const float* C   = (const float*)d_in[4];
    const float* g1  = (const float*)d_in[5];
    const float* b1  = (const float*)d_in[6];
    const float* g2  = (const float*)d_in[7];
    const float* b2  = (const float*)d_in[8];
    const float* Wg1 = (const float*)d_in[9];
    const float* bg1 = (const float*)d_in[10];
    const float* Wg2 = (const float*)d_in[11];
    const float* bg2 = (const float*)d_in[12];
    float* out = (float*)d_out;

    const size_t TOT = 4194304ull;     // 16384 * 256
    const size_t SZB = TOT * 2;        // bf16 buffer bytes
    char* ws = (char*)d_ws;
    __bf16* Xbf   = (__bf16*)(ws);
    __bf16* Xpbf  = (__bf16*)(ws + SZB);
    __bf16* Xcbf  = (__bf16*)(ws + 2 * SZB);
    __bf16* XpT   = (__bf16*)(ws + 3 * SZB);
    __bf16* Wpbf  = (__bf16*)(ws + 4 * SZB);
    __bf16* Wg1bf = Wpbf  + 65536;
    __bf16* Wg2bf = Wg1bf + 65536;
    __bf16* Cblk  = Wg2bf + 65536;
    float*  stats = (float*)(ws + 4 * SZB + 8 * 65536);
    float *sum1 = stats, *ss1 = stats + 256, *sum2 = stats + 512, *ss2 = stats + 768;
    float*  x1    = (float*)(ws + 4 * SZB + 8 * 65536 + 4096);
    float*  xattn = x1 + TOT;
    float*  ybuf  = xattn + TOT;
    __bf16* xbnbf = Xcbf;              // Xcbf dead after attention -> reuse

    zero_kernel<<<4, 256, 0, stream>>>(stats, 1024);

    f2b_kernel<<<(int)(TOT / 256), 256, 0, stream>>>(X, Xbf, (int)TOT);
    f2b_kernel<<<256, 256, 0, stream>>>(Wp,  Wpbf,  65536);
    f2b_kernel<<<256, 256, 0, stream>>>(Wg1, Wg1bf, 65536);
    f2b_kernel<<<256, 256, 0, stream>>>(Wg2, Wg2bf, 65536);
    cblk_kernel<<<256, 256, 0, stream>>>(C, Cblk);

    // Xp = bf16(X @ Wp^T + bp);  Xc = bf16(Xp @ blockdiag(C)^T);  x1 = X @ Wg1^T + bg1
    gemm256_kernel<<<1024, 128, 0, stream>>>(Xbf,  Wpbf,  bp,      nullptr, nullptr, Xpbf, 1);
    gemm256_kernel<<<1024, 128, 0, stream>>>(Xpbf, Cblk,  nullptr, nullptr, nullptr, Xcbf, 1);
    gemm256_kernel<<<1024, 128, 0, stream>>>(Xbf,  Wg1bf, bg1,     nullptr, x1,      nullptr, 0);

    transpose_kernel<<<(int)(TOT / 256), 256, 0, stream>>>(Xpbf, XpT);

    attn_kernel<<<dim3(64, 16), 256, 0, stream>>>(Xpbf, Xcbf, XpT, A, xattn);

    bnstats_kernel<<<64, 256, 0, stream>>>(xattn, sum1, ss1);
    bnapply_kernel<<<(int)(TOT / 256), 256, 0, stream>>>(xattn, sum1, ss1, g1, b1,
                                                         nullptr, xbnbf);

    // x2 = bn1_out @ Wg2^T + bg2, fused sigmoid-gate combine with x1 -> y
    gemm256_kernel<<<1024, 128, 0, stream>>>(xbnbf, Wg2bf, bg2, x1, ybuf, nullptr, 2);

    bnstats_kernel<<<64, 256, 0, stream>>>(ybuf, sum2, ss2);
    bnapply_kernel<<<(int)(TOT / 256), 256, 0, stream>>>(ybuf, sum2, ss2, g2, b2,
                                                         out, nullptr);
}
